// Coarse2FineRefiner_81475529605716
// MI455X (gfx1250) — matrix operand order repaired and checked by reading, not compiled, not run on hardware
//
#include <hip/hip_runtime.h>
#include <hip/hip_bf16.h>

// ---------------- problem constants ----------------
#define BB    128
#define T_C   8
#define T_M   32
#define T_F   128
#define CF    512
#define CMC   128
#define CG    512
#define HID   1024
#define GB2   2048
#define HH    96
#define WWI   96
#define INDIM 640   // CF + CM

// d_out layout (floats): eps_hat | traj_refined | traj_coarse | traj_fine | traj_mid
#define OFF_EPS    0
#define OFF_TRAJR  (BB*T_F*2)             // 32768
#define OFF_TC     (2*BB*T_F*2)           // 65536
#define OFF_TFINE  (OFF_TC + BB*T_C*2)    // 67584
#define OFF_TMID   (OFF_TFINE + BB*T_F*2) // 100352

typedef __attribute__((ext_vector_type(16))) __bf16 v16bf;
typedef __attribute__((ext_vector_type(8)))  float  v8f;

struct bf16x16 { int4 lo, hi; };

__device__ inline unsigned short f2bfu(float f) {
    union { float f; unsigned u; } x; x.f = f;
    unsigned r = x.u + 0x7fffu + ((x.u >> 16) & 1u);   // round-to-nearest-even
    return (unsigned short)(r >> 16);
}
__device__ inline v16bf to_v16bf(int4 lo, int4 hi) {
    bf16x16 t{lo, hi};
    return __builtin_bit_cast(v16bf, t);
}
__device__ inline float siluf(float v) {
    return v * __builtin_amdgcn_rcpf(1.f + __expf(-v));
}
__device__ inline float clip1(float v) { return fminf(fmaxf(v, -1.f), 1.f); }

// ---------------- WMMA fragment loaders (bf16 sources, vector loads only) ----------------
// A: 16x32 (MxK) fragment from bf16 row-major source (row stride lda elems)
__device__ inline v16bf load_a_bf16(const unsigned short* __restrict__ A, int lda, int k0, int lane) {
    int r  = lane & 15;
    int kb = (lane & 16) ? 8 : 0;
    const unsigned short* p = A + (size_t)r * lda + k0 + kb;
    return to_v16bf(*(const int4*)p, *(const int4*)(p + 16));
}
// B: 32x16 (KxN) fragment from pre-packed buffer [tn][kc][lane][16]
__device__ inline v16bf load_b_packed(const unsigned short* __restrict__ Bp, int KC, int tn, int kc, int lane) {
    const unsigned short* p = Bp + (((size_t)tn * KC + kc) * 32 + lane) * 16;
    return to_v16bf(*(const int4*)p, *(const int4*)(p + 8));
}

// ---------------- weight pack: f32 KxN row-major -> bf16 WMMA-B-fragment order ----------------
__global__ void k_pack_b(const float* __restrict__ W, unsigned short* __restrict__ dst, int K, int N) {
    int idx = blockIdx.x * blockDim.x + threadIdx.x;
    if (idx >= K * N) return;
    int KC   = K >> 5;
    int e    = idx & 15;
    int lane = (idx >> 4) & 31;
    int rest = idx >> 9;          // tn*KC + kc
    int kc   = rest % KC;
    int tn   = rest / KC;
    int c    = lane & 15;
    int koff = (lane & 16) ? 8 : 0;
    int k = (kc << 5) + koff + (e < 8 ? e : e + 8);
    int n = (tn << 4) + c;
    dst[idx] = f2bfu(W[(size_t)k * N + n]);
}

__global__ void k_cvt_rows(const float* __restrict__ src, unsigned short* __restrict__ dst, int total) {
    int idx = blockIdx.x * blockDim.x + threadIdx.x;
    if (idx >= total) return;
    dst[idx] = f2bfu(src[idx]);
}

// ---------------- GEMM: out = act(A @ B + bias); one wave per 16x64 tile ----------------
// A bf16 row-major, B packed bf16; writes f32 (Cf) or bf16 row-major (Cbf).
__global__ void k_gemm_bf(const unsigned short* __restrict__ A, const unsigned short* __restrict__ Bp,
                          const float* __restrict__ bias, float* __restrict__ Cf,
                          unsigned short* __restrict__ Cbf,
                          int M, int N, int K, int act) {
    int lane = threadIdx.x & 31;
    int wid  = threadIdx.x >> 5;
    int tile = blockIdx.x * (blockDim.x >> 5) + wid;   // over (M/16)*(N/64)
    int ngrp = N >> 6;
    int tm = tile / ngrp, tg = tile - tm * ngrp;
    int m0  = tm << 4;
    int tn0 = tg << 2;
    int KC  = K >> 5;
    v8f acc[4] = {};
    const unsigned short* Arow = A + (size_t)m0 * K;
    for (int kc = 0; kc < KC; ++kc) {
        v16bf a = load_a_bf16(Arow, K, kc << 5, lane);
#pragma unroll
        for (int j = 0; j < 4; ++j) {
            v16bf b = load_b_packed(Bp, KC, tn0 + j, kc, lane);
            acc[j] = __builtin_amdgcn_wmma_f32_16x16x32_bf16(false, a, false, b, (short)0, acc[j], false, false);
        }
    }
    int rbase = m0 + ((lane & 16) ? 8 : 0);
#pragma unroll
    for (int j = 0; j < 4; ++j) {
        int n = ((tn0 + j) << 4) + (lane & 15);
        float bv = bias[n];
#pragma unroll
        for (int i = 0; i < 8; ++i) {
            float v = acc[j][i] + bv;
            if (act) v = siluf(v);
            if (Cf)  Cf[(size_t)(rbase + i) * N + n] = v;
            else     Cbf[(size_t)(rbase + i) * N + n] = f2bfu(v);
        }
    }
}

// ---------------- FiLM head main: out = silu((xn@W1+b1)*(gamma+1)+beta) @ W2 + b2 ----------------
// one wave per 16-row tile; A staged in LDS; N swept 64 wide; W2 folded in-register.
__global__ void __launch_bounds__(32) k_film(const unsigned short* __restrict__ xn,
                                             const unsigned short* __restrict__ W1p,
                                             const float* __restrict__ b1,
                                             const float* __restrict__ gb,
                                             const float* __restrict__ W2,
                                             const float* __restrict__ b2,
                                             float* __restrict__ out,
                                             int T, int doClip) {
    __shared__ __align__(16) unsigned short sA[16 * INDIM];
    int lane = threadIdx.x;
    int m0 = blockIdx.x << 4;
    {   // cooperative copy: 16 rows x 640 bf16 contiguous = 1280 x int4
        const int4* s4 = (const int4*)(xn + (size_t)m0 * INDIM);
        int4* d4 = (int4*)sA;
        for (int i = lane; i < 1280; i += 32) d4[i] = s4[i];
    }
    __syncthreads();

    const int KC = INDIM >> 5;    // 20
    int rbase = (lane & 16) ? 8 : 0;
    int r  = lane & 15;
    int kb = (lane & 16) ? 8 : 0;
    int bidx[8];
#pragma unroll
    for (int i = 0; i < 8; ++i) bidx[i] = (m0 + rbase + i) / T;
    float oa0[8], oa1[8];
#pragma unroll
    for (int i = 0; i < 8; ++i) { oa0[i] = 0.f; oa1[i] = 0.f; }

    for (int tg = 0; tg < (HID >> 6); ++tg) {   // 16 groups of 64 columns
        v8f acc[4] = {};
        for (int kc = 0; kc < KC; ++kc) {
            const unsigned short* pa = sA + r * INDIM + (kc << 5) + kb;
            v16bf a = to_v16bf(*(const int4*)pa, *(const int4*)(pa + 16));
#pragma unroll
            for (int j = 0; j < 4; ++j) {
                v16bf b = load_b_packed(W1p, KC, (tg << 2) + j, kc, lane);
                acc[j] = __builtin_amdgcn_wmma_f32_16x16x32_bf16(false, a, false, b, (short)0, acc[j], false, false);
            }
        }
#pragma unroll
        for (int j = 0; j < 4; ++j) {
            int n = (((tg << 2) + j) << 4) + (lane & 15);
            float b1v = b1[n];
            float w20 = W2[n * 2 + 0];
            float w21 = W2[n * 2 + 1];
#pragma unroll
            for (int i = 0; i < 8; ++i) {
                const float* gp = gb + (size_t)bidx[i] * GB2 + n;
                float h = (acc[j][i] + b1v) * (gp[0] + 1.f) + gp[HID];
                float s = siluf(h);
                oa0[i] += s * w20;
                oa1[i] += s * w21;
            }
        }
    }
#pragma unroll
    for (int i = 0; i < 8; ++i) {
#pragma unroll
        for (int m = 8; m >= 1; m >>= 1) {
            oa0[i] += __shfl_xor(oa0[i], m, 32);
            oa1[i] += __shfl_xor(oa1[i], m, 32);
        }
    }
    if ((lane & 15) == 0) {
        float bb0 = b2[0], bb1 = b2[1];
#pragma unroll
        for (int i = 0; i < 8; ++i) {
            int row = m0 + rbase + i;
            float v0 = oa0[i] + bb0;
            float v1 = oa1[i] + bb1;
            if (doClip) { v0 = clip1(v0); v1 = clip1(v1); }
            out[row * 2 + 0] = v0;
            out[row * 2 + 1] = v1;
        }
    }
}

// ---------------- time interp (align_corners=True) ----------------
__global__ void k_interp(const float* __restrict__ in, float* __restrict__ out,
                         int Tin, int Tout, int C) {
    int idx = blockIdx.x * blockDim.x + threadIdx.x;
    int total = BB * Tout * C;
    if (idx >= total) return;
    int c = idx % C;
    int t = (idx / C) % Tout;
    int b = idx / (C * Tout);
    float scale = (float)(Tin - 1) / (float)(Tout - 1);
    float pos = t * scale;
    int i0 = (int)floorf(pos);
    int i1 = min(i0 + 1, Tin - 1);
    float w = pos - (float)i0;
    float v0 = in[((size_t)b * Tin + i0) * C + c];
    float v1 = in[((size_t)b * Tin + i1) * C + c];
    out[idx] = v0 * (1.f - w) + v1 * w;
}

// ---------------- bilinear grid sample (align_corners=False, zeros pad) ----------------
__global__ void k_sample_map(const float* __restrict__ mp, const float* __restrict__ traj,
                             float* __restrict__ out, int T) {
    int bt = blockIdx.x;
    int b = bt / T, t = bt - b * T;
    int lane = threadIdx.x;
    float x = traj[((size_t)b * T + t) * 2 + 0];
    float y = traj[((size_t)b * T + t) * 2 + 1];
    float ix = ((x + 1.f) * WWI - 1.f) * 0.5f;
    float iy = ((y + 1.f) * HH  - 1.f) * 0.5f;
    float x0f = floorf(ix), y0f = floorf(iy);
    float wx = ix - x0f, wy = iy - y0f;
    float xs[4] = {x0f, x0f + 1.f, x0f, x0f + 1.f};
    float ys[4] = {y0f, y0f, y0f + 1.f, y0f + 1.f};
    float ws4[4] = {(1.f - wx) * (1.f - wy), wx * (1.f - wy), (1.f - wx) * wy, wx * wy};
    int   cidx[4]; float cw[4];
#pragma unroll
    for (int k = 0; k < 4; ++k) {
        bool valid = (xs[k] >= 0.f) && (xs[k] <= (float)(WWI - 1)) &&
                     (ys[k] >= 0.f) && (ys[k] <= (float)(HH - 1));
        int xi = min(max((int)xs[k], 0), WWI - 1);
        int yi = min(max((int)ys[k], 0), HH - 1);
        cidx[k] = yi * WWI + xi;
        cw[k] = valid ? ws4[k] : 0.f;
    }
    const float* mb = mp + (size_t)b * CMC * HH * WWI;
    for (int ch = lane; ch < CMC; ch += 32) {
        const float* mc = mb + (size_t)ch * HH * WWI;
        float acc = 0.f;
#pragma unroll
        for (int k = 0; k < 4; ++k) acc += mc[cidx[k]] * cw[k];
        out[((size_t)(b * T + t)) * CMC + ch] = acc;
    }
}

// ---------------- concat(feat, map) + LayerNorm -> bf16 rows ----------------
__global__ void __launch_bounds__(32) k_concat_ln(const float* __restrict__ feat,
                                                  const float* __restrict__ mp,
                                                  const float* __restrict__ ln_g,
                                                  const float* __restrict__ ln_b,
                                                  unsigned short* __restrict__ xn) {
    int row = blockIdx.x;
    int lane = threadIdx.x;
    const float* fr = feat + (size_t)row * CF;
    const float* mr = mp   + (size_t)row * CMC;
    float xv[20];
    float s = 0.f, s2 = 0.f;
#pragma unroll
    for (int i = 0; i < 20; ++i) {
        int e = lane + 32 * i;
        float v = (e < CF) ? fr[e] : mr[e - CF];
        xv[i] = v; s += v; s2 += v * v;
    }
#pragma unroll
    for (int m = 16; m >= 1; m >>= 1) { s += __shfl_xor(s, m, 32); s2 += __shfl_xor(s2, m, 32); }
    float mu  = s / (float)INDIM;
    float var = s2 / (float)INDIM - mu * mu;
    float inv = rsqrtf(var + 1e-5f);
    unsigned short* xr = xn + (size_t)row * INDIM;
#pragma unroll
    for (int i = 0; i < 20; ++i) {
        int e = lane + 32 * i;
        float v = (xv[i] - mu) * inv * ln_g[e] + ln_b[e];
        xr[e] = f2bfu(v);
    }
}

// ---------------- traj_fine = clip(interp(res_mid,128) + interp(traj_coarse,128)) ----------------
__global__ void k_trajfine(const float* __restrict__ res_mid, const float* __restrict__ traj_c,
                           float* __restrict__ out) {
    int idx = blockIdx.x * blockDim.x + threadIdx.x;
    if (idx >= BB * T_F * 2) return;
    int c = idx & 1;
    int t = (idx >> 1) % T_F;
    int b = idx / (2 * T_F);
    float p1 = t * (31.f / 127.f);
    int i0 = (int)floorf(p1); int i1 = min(i0 + 1, 31);
    float w1 = p1 - (float)i0;
    float v1 = res_mid[(b * 32 + i0) * 2 + c] * (1.f - w1) + res_mid[(b * 32 + i1) * 2 + c] * w1;
    float p2 = t * (7.f / 127.f);
    int j0 = (int)floorf(p2); int j1 = min(j0 + 1, 7);
    float w2 = p2 - (float)j0;
    float v2 = traj_c[(b * 8 + j0) * 2 + c] * (1.f - w2) + traj_c[(b * 8 + j1) * 2 + c] * w2;
    out[idx] = clip1(v1 + v2);
}

__global__ void k_trajmid_aux(const float* __restrict__ res_mid, const float* __restrict__ traj_mid,
                              float* __restrict__ out) {
    int idx = blockIdx.x * blockDim.x + threadIdx.x;
    if (idx >= BB * T_M * 2) return;
    out[idx] = clip1(res_mid[idx] + traj_mid[idx]);
}

// ---------------- final: traj_refined + eps_hat + aux traj_fine ----------------
__global__ void k_final(const float* __restrict__ x_t, const int* __restrict__ ts,
                        const float* __restrict__ alphas, const float* __restrict__ traj_fine,
                        const float* __restrict__ res_fine, float* __restrict__ dout) {
    int idx = blockIdx.x * blockDim.x + threadIdx.x;   // over B*T_F
    if (idx >= BB * T_F) return;
    int b = idx / T_F;
    float ab  = alphas[ts[b]];
    float sab = sqrtf(ab);
    float som = sqrtf(1.f - ab);
#pragma unroll
    for (int c = 0; c < 2; ++c) {
        int o = idx * 2 + c;
        float tf = clip1(traj_fine[o] + res_fine[o]);
        float eps = (x_t[o] - sab * tf) / (som + 1e-8f);
        dout[OFF_EPS   + o] = eps;
        dout[OFF_TRAJR + o] = tf;
        dout[OFF_TFINE + o] = tf;
    }
}

// ---------------- host launcher ----------------
struct HeadP { const float *W1,*b1,*Wg1,*bg1,*Wg2,*bg2,*W2,*b2,*ln_g,*ln_b; };

static inline int cdiv(int a, int b) { return (a + b - 1) / b; }

extern "C" void kernel_launch(void* const* d_in, const int* in_sizes, int n_in,
                              void* d_out, int out_size, void* d_ws, size_t ws_size,
                              hipStream_t stream) {
    (void)n_in; (void)out_size; (void)ws_size;
    const float* in_feat     = (const float*)d_in[0];
    const float* map_cond    = (const float*)d_in[1];
    const float* global_cond = (const float*)d_in[2];
    const float* x_t_in      = (const float*)d_in[3];
    const int*   timesteps   = (const int*)d_in[4];
    const float* alphas      = (const float*)d_in[5];

    // Params may flatten sorted (jax tree) or in insertion order; disambiguate by size.
    bool sorted_order = (in_sizes[6] != 640);
    auto mk = [&](int base) {
        HeadP h;
        if (sorted_order) { // W1, W2, Wg1, Wg2, b1, b2, bg1, bg2, ln_b, ln_g
            h.W1   = (const float*)d_in[base + 0];
            h.W2   = (const float*)d_in[base + 1];
            h.Wg1  = (const float*)d_in[base + 2];
            h.Wg2  = (const float*)d_in[base + 3];
            h.b1   = (const float*)d_in[base + 4];
            h.b2   = (const float*)d_in[base + 5];
            h.bg1  = (const float*)d_in[base + 6];
            h.bg2  = (const float*)d_in[base + 7];
            h.ln_b = (const float*)d_in[base + 8];
            h.ln_g = (const float*)d_in[base + 9];
        } else {            // ln_g, ln_b, W1, b1, Wg1, bg1, Wg2, bg2, W2, b2
            h.ln_g = (const float*)d_in[base + 0];
            h.ln_b = (const float*)d_in[base + 1];
            h.W1   = (const float*)d_in[base + 2];
            h.b1   = (const float*)d_in[base + 3];
            h.Wg1  = (const float*)d_in[base + 4];
            h.bg1  = (const float*)d_in[base + 5];
            h.Wg2  = (const float*)d_in[base + 6];
            h.bg2  = (const float*)d_in[base + 7];
            h.W2   = (const float*)d_in[base + 8];
            h.b2   = (const float*)d_in[base + 9];
        }
        return h;
    };
    HeadP heads[3] = { mk(6), mk(16), mk(26) };

    // workspace layout (byte offsets)
    char* ws = (char*)d_ws;
    float* h_coarse  = (float*)(ws + 0);          // 2,097,152
    float* map_c     = (float*)(ws + 2097152);    //   524,288
    float* h_mid     = (float*)(ws + 2621440);    // 8,388,608
    float* map_m     = (float*)(ws + 11010048);   // 2,097,152
    float* map_f     = (float*)(ws + 13107200);   // 8,388,608
    unsigned short* tmp_g1bf = (unsigned short*)(ws + 21495808); // 524,288
    float* gbuf      = (float*)(ws + 22020096);   // 1,048,576
    unsigned short* xn_c = (unsigned short*)(ws + 23068672); // 1,310,720
    unsigned short* xn_m = (unsigned short*)(ws + 24379392); // 5,242,880
    unsigned short* xn_f = (unsigned short*)(ws + 29622272); // 20,971,520
    float* init_c    = (float*)(ws + 50593792);   // 8,192
    float* traj_mid  = (float*)(ws + 50601984);   // 32,768
    float* res_mid   = (float*)(ws + 50634752);   // 32,768
    float* traj_fine = (float*)(ws + 50667520);   // 131,072
    float* res_fine  = (float*)(ws + 50798592);   // 131,072
    unsigned short* cond_bf = (unsigned short*)(ws + 50929664); // 131,072
    unsigned short* pWg1 = (unsigned short*)(ws + 51060736);    // 2,097,152
    unsigned short* pWg2 = (unsigned short*)(ws + 53157888);    // 8,388,608
    unsigned short* pW1  = (unsigned short*)(ws + 61546496);    // 1,310,720

    float* out = (float*)d_out;
    float* traj_coarse = out + OFF_TC;

    const int gemm_blocks = ((BB / 16) * (GB2 / 64)) / 4; // 256 waves, 4 waves/block

    // one-time activation conversion
    k_cvt_rows<<<cdiv(BB * CG, 256), 256, 0, stream>>>(global_cond, cond_bf, BB * CG);

    // per-head cond-GEMM pipeline + FiLM main
    auto run_head = [&](const HeadP& h, const unsigned short* xn, float* film_out, int T, int doClip) {
        k_pack_b<<<cdiv(CG * GB2, 256), 256, 0, stream>>>(h.Wg1, pWg1, CG, GB2);
        k_gemm_bf<<<gemm_blocks, 128, 0, stream>>>(cond_bf, pWg1, h.bg1, nullptr, tmp_g1bf, BB, GB2, CG, 1);
        k_pack_b<<<cdiv(GB2 * GB2, 256), 256, 0, stream>>>(h.Wg2, pWg2, GB2, GB2);
        k_gemm_bf<<<gemm_blocks, 128, 0, stream>>>(tmp_g1bf, pWg2, h.bg2, gbuf, nullptr, BB, GB2, GB2, 0);
        k_pack_b<<<cdiv(INDIM * HID, 256), 256, 0, stream>>>(h.W1, pW1, INDIM, HID);
        k_film<<<(BB * T) / 16, 32, 0, stream>>>(xn, pW1, h.b1, gbuf, h.W2, h.b2, film_out, T, doClip);
    };

    // ---------------- coarse stage ----------------
    k_interp<<<cdiv(BB * T_C * 2, 256), 256, 0, stream>>>(x_t_in, init_c, T_F, T_C, 2);
    k_interp<<<cdiv(BB * T_C * CF, 256), 256, 0, stream>>>(in_feat, h_coarse, T_F, T_C, CF);
    k_sample_map<<<BB * T_C, 32, 0, stream>>>(map_cond, init_c, map_c, T_C);
    k_concat_ln<<<BB * T_C, 32, 0, stream>>>(h_coarse, map_c, heads[0].ln_g, heads[0].ln_b, xn_c);
    run_head(heads[0], xn_c, traj_coarse, T_C, 1);

    // ---------------- mid stage ----------------
    k_interp<<<cdiv(BB * T_M * 2, 256), 256, 0, stream>>>(traj_coarse, traj_mid, T_C, T_M, 2);
    k_interp<<<cdiv(BB * T_M * CF, 256), 256, 0, stream>>>(in_feat, h_mid, T_F, T_M, CF);
    k_sample_map<<<BB * T_M, 32, 0, stream>>>(map_cond, traj_mid, map_m, T_M);
    k_concat_ln<<<BB * T_M, 32, 0, stream>>>(h_mid, map_m, heads[1].ln_g, heads[1].ln_b, xn_m);
    run_head(heads[1], xn_m, res_mid, T_M, 0);
    k_trajmid_aux<<<cdiv(BB * T_M * 2, 256), 256, 0, stream>>>(res_mid, traj_mid, out + OFF_TMID);

    // ---------------- fine stage ----------------
    k_trajfine<<<cdiv(BB * T_F * 2, 256), 256, 0, stream>>>(res_mid, traj_coarse, traj_fine);
    k_sample_map<<<BB * T_F, 32, 0, stream>>>(map_cond, traj_fine, map_f, T_F);
    k_concat_ln<<<BB * T_F, 32, 0, stream>>>(in_feat, map_f, heads[2].ln_g, heads[2].ln_b, xn_f);
    run_head(heads[2], xn_f, res_fine, T_F, 0);
    k_final<<<cdiv(BB * T_F, 256), 256, 0, stream>>>(x_t_in, timesteps, alphas, traj_fine, res_fine, out);
}
